// TypeNet_74586402063303
// MI455X (gfx1250) — matrix operand order repaired and checked by reading, not compiled, not run on hardware
//
#include <hip/hip_runtime.h>

// ---- problem constants (TypeNet) ----
#define T_   128
#define H_   128
#define B_   1024
#define EPS_ 1e-5f
#define MARGIN_ 1.5f
#define BTH_ ((size_t)B_ * T_ * H_)

// fragmentized weight matrix: 32 ntiles * 4 ksteps * 32 lanes * 16 halves
#define FRAG_HALVES 65536
#define FRAG_BYTES  (FRAG_HALVES * 2)
#define HLDS_HALVES (16 * 136)        // h exchange tile, padded stride

typedef __attribute__((ext_vector_type(16))) _Float16 v16h;
typedef __attribute__((ext_vector_type(8)))  _Float16 v8h;
typedef __attribute__((ext_vector_type(8)))  float    v8f;
typedef __attribute__((ext_vector_type(4)))  float    v4f;

union V16H { v16h v; v8h h[2]; };
union V8F  { v8f  v; v4f f[2]; };

__device__ __forceinline__ float sigm_(float x) { return 1.f / (1.f + __expf(-x)); }

__device__ __forceinline__ v8f wmma_f16_(v16h a, v16h b, v8f c) {
  // D = A(16x32 f16) x B(32x16 f16) + C(16x16 f32)
  return __builtin_amdgcn_wmma_f32_16x16x32_f16(
      /*neg_a=*/false, a, /*neg_b=*/false, b,
      /*c_mod=*/(short)0, c, /*reuse_a=*/false, /*reuse_b=*/false);
}

// -----------------------------------------------------------------------------
// Rewrite W[512][128] (row-major f32; W^T is the 128x512 WMMA B matrix) into the
// native CDNA5 B-fragment layout (ISA 7.12.2):
//   frag[((nt*4 + ks)*32 + lane)*16 + e] = (f16) W[n][k]
//   n = nt*16 + lane%16 ; k = ks*32 + (lane>=16 ? 8 : 0) + (e&7) + (e>=8 ? 16 : 0)
// -----------------------------------------------------------------------------
__global__ void k_fragmentize(const float* __restrict__ W, _Float16* __restrict__ frag) {
  int id   = blockIdx.x * blockDim.x + threadIdx.x;   // 65536 threads
  int e    = id & 15;
  int lane = (id >> 4) & 31;
  int ks   = (id >> 9) & 3;
  int nt   = id >> 11;
  int n = nt * 16 + (lane & 15);
  int k = ks * 32 + ((lane >> 4) << 3) + (e & 7) + ((e >> 3) << 4);
  frag[id] = (_Float16)W[n * H_ + k];
}

// -----------------------------------------------------------------------------
// BN over timestep channels (channel = t, stats over B x D=3); writes the
// normalized input transposed to [T][3][B] for batch-contiguous scan loads.
// grid = (T_, 3 branches)
// -----------------------------------------------------------------------------
__global__ void k_bn1(const float* __restrict__ x0, const float* __restrict__ x1,
                      const float* __restrict__ x2,
                      const float* __restrict__ gamma, const float* __restrict__ beta,
                      float* __restrict__ xbn_all) {
  const int t  = blockIdx.x;
  const int br = blockIdx.y;
  const float* x = (br == 0) ? x0 : (br == 1) ? x1 : x2;
  float* xbn_t = xbn_all + (size_t)br * T_ * 3 * B_;

  __shared__ float s0[256], s1[256];
  float sum = 0.f, sq = 0.f;
  for (int b = threadIdx.x; b < B_; b += 256) {
    const float* p = x + ((size_t)b * T_ + t) * 3;
    #pragma unroll
    for (int d = 0; d < 3; ++d) { float v = p[d]; sum += v; sq += v * v; }
  }
  s0[threadIdx.x] = sum; s1[threadIdx.x] = sq;
  __syncthreads();
  for (int s = 128; s > 0; s >>= 1) {
    if (threadIdx.x < s) { s0[threadIdx.x] += s0[threadIdx.x + s]; s1[threadIdx.x] += s1[threadIdx.x + s]; }
    __syncthreads();
  }
  const float inv  = 1.f / (float)(B_ * 3);
  const float mean = s0[0] * inv;
  const float var  = s1[0] * inv - mean * mean;
  const float sc   = gamma[t] * rsqrtf(var + EPS_);
  const float sh   = beta[t] - mean * sc;
  for (int b = threadIdx.x; b < B_; b += 256) {
    const float* p = x + ((size_t)b * T_ + t) * 3;
    #pragma unroll
    for (int d = 0; d < 3; ++d)
      xbn_t[((size_t)t * 3 + d) * B_ + b] = p[d] * sc + sh;
  }
}

// -----------------------------------------------------------------------------
// BN over feature channels (channel = h, stats over B x T), f16 output
// (WMMA A operand of the fused layer-2 input projection).
// grid = (H_, 3 branches)
// -----------------------------------------------------------------------------
__global__ void k_bn2(const float* __restrict__ y_all,
                      const float* __restrict__ gamma, const float* __restrict__ beta,
                      _Float16* __restrict__ abn_all) {
  const int c  = blockIdx.x;
  const int br = blockIdx.y;
  const float* y = y_all + (size_t)br * BTH_;
  _Float16* abn  = abn_all + (size_t)br * BTH_;

  __shared__ float s0[256], s1[256];
  float sum = 0.f, sq = 0.f;
  for (int i = threadIdx.x; i < B_ * T_; i += 256) {
    float v = y[(size_t)i * H_ + c]; sum += v; sq += v * v;
  }
  s0[threadIdx.x] = sum; s1[threadIdx.x] = sq;
  __syncthreads();
  for (int s = 128; s > 0; s >>= 1) {
    if (threadIdx.x < s) { s0[threadIdx.x] += s0[threadIdx.x + s]; s1[threadIdx.x] += s1[threadIdx.x + s]; }
    __syncthreads();
  }
  const float inv  = 1.f / (float)(B_ * T_);
  const float mean = s0[0] * inv;
  const float var  = s1[0] * inv - mean * mean;
  const float sc   = gamma[c] * rsqrtf(var + EPS_);
  const float sh   = beta[c] - mean * sc;
  for (int i = threadIdx.x; i < B_ * T_; i += 256)
    abn[(size_t)i * H_ + c] = (_Float16)(y[(size_t)i * H_ + c] * sc + sh);
}

// -----------------------------------------------------------------------------
// LSTM layer 1 scan. One WG = 16 batch rows; 8 waves, wave w owns hidden cols
// [16w,16w+16) and its i/f/g/o gate tiles (ntiles w,w+8,w+16,w+24) so the gate
// math + cell state live in registers. Whh fragments are staged in LDS (128KB)
// so per-step operand fetches are ds_load_b128, not scratch/global. h exchanged
// through a 4.25KB LDS tile with 2 barriers/step.
// dynamic LDS: [0,128KB) whh frags, then h tile.  grid = (B_/16, 3)
// -----------------------------------------------------------------------------
__global__ void __launch_bounds__(256)
k_lstm1(const float* __restrict__ xbn_all, const float* __restrict__ Wih1,
        const float* __restrict__ bih, const float* __restrict__ bhh,
        const _Float16* __restrict__ whhf, float* __restrict__ y_all) {
  extern __shared__ __align__(16) _Float16 smem[];
  _Float16* wlds = smem;                       // FRAG_HALVES
  _Float16* hlds = smem + FRAG_HALVES;         // HLDS_HALVES

  const int br    = blockIdx.y;
  const float* xbn_t = xbn_all + (size_t)br * T_ * 3 * B_;
  float* yout        = y_all   + (size_t)br * BTH_;

  const int btile = blockIdx.x;
  const int lane  = threadIdx.x & 31;
  const int w     = threadIdx.x >> 5;
  const int m     = lane & 15;              // A row / C column selector
  const int kofs  = (lane >> 4) << 3;       // A-frag K base (0/8) per ISA layout
  const int rofs  = kofs;                   // C/D rows (0..7 / 8..15)
  const int hcol  = w * 16 + m;
  const int bbase = btile * 16;

  { // stage Whh fragments into LDS (one 128KB coop copy)
    const v8h* src = (const v8h*)whhf;
    v8h*       dst = (v8h*)wlds;
    for (int i = threadIdx.x; i < FRAG_HALVES / 8; i += 256) dst[i] = src[i];
  }
  for (int i = threadIdx.x; i < HLDS_HALVES; i += 256) hlds[i] = (_Float16)0.f;

  float bsum[4], wc[4][3];
  #pragma unroll
  for (int g4 = 0; g4 < 4; ++g4) {
    int g = g4 * H_ + hcol;
    bsum[g4] = bih[g] + bhh[g];
    #pragma unroll
    for (int d = 0; d < 3; ++d) wc[g4][d] = Wih1[g * 3 + d];
  }

  const int fbase = w * 2048 + lane * 16;   // per-lane frag base (halves)
  const v8f vzero = {0.f,0.f,0.f,0.f,0.f,0.f,0.f,0.f};
  v8f c = vzero;

  for (int t = 0; t < T_; ++t) {
    __syncthreads();                        // h(t-1) + (t==0) staged weights visible
    v16h a[4];
    #pragma unroll
    for (int ks = 0; ks < 4; ++ks) {
      const _Float16* p = &hlds[m * 136 + ks * 32 + kofs];
      V16H u; u.h[0] = *(const v8h*)p; u.h[1] = *(const v8h*)(p + 16);
      a[ks] = u.v;
    }
    __syncthreads();                        // reads done before h(t) writes

    int fb = fbase;
    asm volatile("" : "+v"(fb));            // opaque per-iteration: keep ds loads
                                            // in-loop (no hoist -> no spill)
    v8f acc[4] = {vzero, vzero, vzero, vzero};
    #pragma unroll
    for (int ks = 0; ks < 4; ++ks) {
      #pragma unroll
      for (int g4 = 0; g4 < 4; ++g4) {
        const v16h b = *(const v16h*)&wlds[fb + g4 * 16384 + ks * 512];
        acc[g4] = wmma_f16_(a[ks], b, acc[g4]);
      }
    }

    // fused input projection: G += xbn[b,t,:] @ Wih1^T  (D = 3)
    V8F xd[3];
    #pragma unroll
    for (int d = 0; d < 3; ++d) {
      const float* p = xbn_t + ((size_t)t * 3 + d) * B_ + bbase + rofs;
      xd[d].f[0] = *(const v4f*)p;
      xd[d].f[1] = *(const v4f*)(p + 4);
    }

    #pragma unroll
    for (int v = 0; v < 8; ++v) {
      float gi = acc[0][v] + bsum[0];
      float gf = acc[1][v] + bsum[1];
      float gg = acc[2][v] + bsum[2];
      float go = acc[3][v] + bsum[3];
      #pragma unroll
      for (int d = 0; d < 3; ++d) {
        float xv = xd[d].v[v];
        gi += xv * wc[0][d]; gf += xv * wc[1][d];
        gg += xv * wc[2][d]; go += xv * wc[3][d];
      }
      float ii = sigm_(gi), ff = sigm_(gf), gt = tanhf(gg), oo = sigm_(go);
      float cv = ff * c[v] + ii * gt;
      c[v] = cv;
      float hv = oo * tanhf(cv);
      hlds[(rofs + v) * 136 + hcol] = (_Float16)hv;
      yout[(((size_t)(bbase + rofs + v)) * T_ + t) * H_ + hcol] = hv;
    }
  }
}

// -----------------------------------------------------------------------------
// LSTM layer 2 scan with fused input projection. Both Whh2 and Wih2 fragment
// sets staged in LDS (256KB total, fits the 320KB WGP budget); per timestep each
// wave runs two WMMA accumulation chains into the same C tile:
// A2(bn2 out, f16, global) x Wih2^T  then  h(t-1) x Whh2^T.
// dynamic LDS: [0,128KB) whh frags, [128KB,256KB) wih frags, then h tile.
// grid = (B_/16, 3); writes directly into d_out.
// -----------------------------------------------------------------------------
__global__ void __launch_bounds__(256)
k_lstm2(const _Float16* __restrict__ abn_all, const _Float16* __restrict__ wihf,
        const float* __restrict__ bih, const float* __restrict__ bhh,
        const _Float16* __restrict__ whhf, float* __restrict__ out_all) {
  extern __shared__ __align__(16) _Float16 smem[];
  _Float16* wlds_h = smem;                          // FRAG_HALVES (Whh2)
  _Float16* wlds_i = smem + FRAG_HALVES;            // FRAG_HALVES (Wih2)
  _Float16* hlds   = smem + 2 * FRAG_HALVES;        // HLDS_HALVES

  const int br = blockIdx.y;
  const _Float16* abn = abn_all + (size_t)br * BTH_;
  float* yout         = out_all + (size_t)br * BTH_;

  const int btile = blockIdx.x;
  const int lane  = threadIdx.x & 31;
  const int w     = threadIdx.x >> 5;
  const int m     = lane & 15;
  const int kofs  = (lane >> 4) << 3;
  const int rofs  = kofs;
  const int hcol  = w * 16 + m;
  const int bbase = btile * 16;

  { // stage both fragment sets into LDS
    const v8h* s0 = (const v8h*)whhf;
    const v8h* s1 = (const v8h*)wihf;
    v8h* d0 = (v8h*)wlds_h;
    v8h* d1 = (v8h*)wlds_i;
    for (int i = threadIdx.x; i < FRAG_HALVES / 8; i += 256) { d0[i] = s0[i]; d1[i] = s1[i]; }
  }
  for (int i = threadIdx.x; i < HLDS_HALVES; i += 256) hlds[i] = (_Float16)0.f;

  float bsum[4];
  #pragma unroll
  for (int g4 = 0; g4 < 4; ++g4) bsum[g4] = bih[g4 * H_ + hcol] + bhh[g4 * H_ + hcol];

  const _Float16* arow0 = abn + ((size_t)(bbase + m) * T_) * H_;  // A2 row for this lane
  const int fbase = w * 2048 + lane * 16;
  const v8f vzero = {0.f,0.f,0.f,0.f,0.f,0.f,0.f,0.f};
  v8f c = vzero;

  for (int t = 0; t < T_; ++t) {
    __syncthreads();
    v16h a[4];
    #pragma unroll
    for (int ks = 0; ks < 4; ++ks) {
      const _Float16* p = &hlds[m * 136 + ks * 32 + kofs];
      V16H u; u.h[0] = *(const v8h*)p; u.h[1] = *(const v8h*)(p + 16);
      a[ks] = u.v;
    }
    __syncthreads();

    const _Float16* arow = arow0 + (size_t)t * H_;
    __builtin_prefetch(arow + H_, 0, 0);            // next timestep's A2 operand
    v16h a2[4];
    #pragma unroll
    for (int ks = 0; ks < 4; ++ks) {
      const _Float16* p = arow + ks * 32 + kofs;
      V16H u; u.h[0] = *(const v8h*)p; u.h[1] = *(const v8h*)(p + 16);
      a2[ks] = u.v;
    }

    int fb = fbase;
    asm volatile("" : "+v"(fb));            // keep the LDS frag loads in-loop
    v8f acc[4] = {vzero, vzero, vzero, vzero};
    #pragma unroll
    for (int ks = 0; ks < 4; ++ks) {
      #pragma unroll
      for (int g4 = 0; g4 < 4; ++g4) {
        const int fo = fb + g4 * 16384 + ks * 512;
        const v16h bi = *(const v16h*)&wlds_i[fo];
        const v16h bh = *(const v16h*)&wlds_h[fo];
        acc[g4] = wmma_f16_(a2[ks], bi, acc[g4]);   // input projection
        acc[g4] = wmma_f16_(a[ks],  bh, acc[g4]);   // recurrence
      }
    }

    #pragma unroll
    for (int v = 0; v < 8; ++v) {
      float gi = acc[0][v] + bsum[0];
      float gf = acc[1][v] + bsum[1];
      float gg = acc[2][v] + bsum[2];
      float go = acc[3][v] + bsum[3];
      float ii = sigm_(gi), ff = sigm_(gf), gt = tanhf(gg), oo = sigm_(go);
      float cv = ff * c[v] + ii * gt;
      c[v] = cv;
      float hv = oo * tanhf(cv);
      hlds[(rofs + v) * 136 + hcol] = (_Float16)hv;
      yout[(((size_t)(bbase + rofs + v)) * T_ + t) * H_ + hcol] = hv;
    }
  }
}

// -----------------------------------------------------------------------------
// Triplet loss. Stage 1: per-batch squared distances. Stage 2: mean of hinge.
// -----------------------------------------------------------------------------
__global__ void k_loss_partial(const float* __restrict__ a, const float* __restrict__ p,
                               const float* __restrict__ n,
                               float* __restrict__ dp, float* __restrict__ dn) {
  const int b = blockIdx.x;
  __shared__ float s0[256], s1[256];
  const float* ab = a + (size_t)b * T_ * H_;
  const float* pb = p + (size_t)b * T_ * H_;
  const float* nb = n + (size_t)b * T_ * H_;
  float sp = 0.f, sn = 0.f;
  for (int i = threadIdx.x; i < T_ * H_; i += 256) {
    float av = ab[i];
    float d1 = av - pb[i]; sp += d1 * d1;
    float d2 = av - nb[i]; sn += d2 * d2;
  }
  s0[threadIdx.x] = sp; s1[threadIdx.x] = sn;
  __syncthreads();
  for (int s = 128; s > 0; s >>= 1) {
    if (threadIdx.x < s) { s0[threadIdx.x] += s0[threadIdx.x + s]; s1[threadIdx.x] += s1[threadIdx.x + s]; }
    __syncthreads();
  }
  if (threadIdx.x == 0) { dp[b] = s0[0]; dn[b] = s1[0]; }
}

__global__ void k_loss_final(const float* __restrict__ dp, const float* __restrict__ dn,
                             float* __restrict__ loss) {
  __shared__ float s0[256];
  float s = 0.f;
  for (int b = threadIdx.x; b < B_; b += 256) {
    float v = dp[b] - dn[b] + MARGIN_;
    s += (v > 0.f) ? v : 0.f;
  }
  s0[threadIdx.x] = s;
  __syncthreads();
  for (int st = 128; st > 0; st >>= 1) {
    if (threadIdx.x < st) s0[threadIdx.x] += s0[threadIdx.x + st];
    __syncthreads();
  }
  if (threadIdx.x == 0) loss[0] = s0[0] * (1.f / (float)B_);
}

// -----------------------------------------------------------------------------
extern "C" void kernel_launch(void* const* d_in, const int* in_sizes, int n_in,
                              void* d_out, int out_size, void* d_ws, size_t ws_size,
                              hipStream_t stream) {
  (void)in_sizes; (void)n_in; (void)out_size; (void)ws_size;
  const float* anchor   = (const float*)d_in[0];
  const float* positive = (const float*)d_in[1];
  const float* negative = (const float*)d_in[2];
  const float* W_ih1 = (const float*)d_in[3];
  const float* W_hh1 = (const float*)d_in[4];
  const float* b_ih1 = (const float*)d_in[5];
  const float* b_hh1 = (const float*)d_in[6];
  const float* W_ih2 = (const float*)d_in[7];
  const float* W_hh2 = (const float*)d_in[8];
  const float* b_ih2 = (const float*)d_in[9];
  const float* b_hh2 = (const float*)d_in[10];
  const float* bn1_g = (const float*)d_in[11];
  const float* bn1_b = (const float*)d_in[12];
  const float* bn2_g = (const float*)d_in[13];
  const float* bn2_b = (const float*)d_in[14];

  // workspace layout (~293 MiB)
  char* ws = (char*)d_ws;
  float*    y1_all   = (float*)ws;                                    // 3*B*T*H f32 (192MiB)
  _Float16* abn_all  = (_Float16*)(ws + 3 * BTH_ * sizeof(float));    // 3*B*T*H f16 (96MiB)
  float*    xbn_all  = (float*)((char*)abn_all + 3 * BTH_ * sizeof(_Float16)); // 3*T*3*B f32
  _Float16* whh1f    = (_Float16*)((char*)xbn_all + (size_t)3 * T_ * 3 * B_ * sizeof(float));
  _Float16* whh2f    = whh1f + FRAG_HALVES;
  _Float16* wih2f    = whh2f + FRAG_HALVES;
  float*    dp       = (float*)(wih2f + FRAG_HALVES);
  float*    dn       = dp + B_;

  // one-time weight fragmentization into WMMA B-operand layout
  k_fragmentize<<<256, 256, 0, stream>>>(W_hh1, whh1f);
  k_fragmentize<<<256, 256, 0, stream>>>(W_hh2, whh2f);
  k_fragmentize<<<256, 256, 0, stream>>>(W_ih2, wih2f);

  float* out = (float*)d_out;

  const size_t lds1 = (FRAG_BYTES)     + HLDS_HALVES * sizeof(_Float16); // 135,424 B
  const size_t lds2 = (2 * FRAG_BYTES) + HLDS_HALVES * sizeof(_Float16); // 266,496 B

  // all three branches processed concurrently via grid.y
  k_bn1  <<<dim3(T_, 3),      256, 0,    stream>>>(anchor, positive, negative, bn1_g, bn1_b, xbn_all);
  k_lstm1<<<dim3(B_ / 16, 3), 256, lds1, stream>>>(xbn_all, W_ih1, b_ih1, b_hh1, whh1f, y1_all);
  k_bn2  <<<dim3(H_, 3),      256, 0,    stream>>>(y1_all, bn2_g, bn2_b, abn_all);
  k_lstm2<<<dim3(B_ / 16, 3), 256, lds2, stream>>>(abn_all, wih2f, b_ih2, b_hh2, whh2f, out);

  k_loss_partial<<<B_, 256, 0, stream>>>(out, out + BTH_, out + 2 * BTH_, dp, dn);
  k_loss_final  <<<1,   256, 0, stream>>>(dp, dn, out + 3 * BTH_);
}
